// MACE_76441827934646
// MI455X (gfx1250) — matrix operand order repaired and checked by reading, not compiled, not run on hardware
//
// MACE-style GNN forward for MI455X (gfx1250, wave32, WMMA).
// Dense GEMMs -> v_wmma_f32_16x16x32_bf16 (bf16 operands, f32 acc), 2 WMMAs
// per wave per K-step (32x64 block tile). A tile fed by async global->LDS
// copies (global_load_async_to_lds_b128, ASYNCcnt + s_wait_asynccnt) via
// inline asm on gfx1250; B staged TRANSPOSED in LDS with 36-float pitch so
// every fragment is 2-4 contiguous ds_load_b128 (conflict-free banks).
// Interior tiles use branchless float4 global loads; edges fall back to
// guarded loads. segment_sum via global f32 atomics; next K-tile prefetched
// via global_prefetch_b8. Workspace ~700 MB.
#include <hip/hip_runtime.h>
#include <hip/hip_bf16.h>

typedef __attribute__((ext_vector_type(16))) __bf16 v16bf;
typedef __attribute__((ext_vector_type(8)))  float  v8f;

#define OUTC 1604
#define BM 32
#define BN 64

#if defined(__gfx1250__)
#define HAVE_ASYNC_LDS 1
#else
#define HAVE_ASYNC_LDS 0
#endif

#if HAVE_ASYNC_LDS
// 16-byte global->LDS async copy (GLOBAL_LOAD_ASYNC_TO_LDS_B128, ASYNCcnt).
// VGLOBAL GV mode: vdst = LDS byte address VGPR, vaddr = 64-bit global addr.
__device__ __forceinline__ void async_g2l_16B(const float* gsrc, float* ldst) {
  unsigned loff = (unsigned)(unsigned long long)ldst;   // low 32 bits = LDS offset
  unsigned long long gaddr = (unsigned long long)gsrc;
  asm volatile("global_load_async_to_lds_b128 %0, %1, off"
               :: "v"(loff), "v"(gaddr)
               : "memory");
}
__device__ __forceinline__ void wait_async0() {
  asm volatile("s_wait_asynccnt 0x0" ::: "memory");
}
#endif

__constant__ int d_LMAP[16] = {0,1,1,1,2,2,2,2,2,3,3,3,3,3,3,3};

__device__ __forceinline__ float act_apply(float v, int act, float a) {
  switch (act) {
    case 1: return v / (1.0f + __expf(-v));          // silu
    case 2: return 1.0f / (1.0f + __expf(-v));       // sigmoid
    case 3: return 0.5f * (tanhf(v) + 1.0f);         // 0.5*(tanh+1)
    case 4: return v >= 0.0f ? v : a * v;            // prelu
    case 5: return fmaxf(v, 0.0f);                   // relu
    default: return v;
  }
}

__device__ __forceinline__ v16bf pack16(float4 a, float4 b, float4 c, float4 d) {
  v16bf r;
  r[0]  = (__bf16)a.x; r[1]  = (__bf16)a.y; r[2]  = (__bf16)a.z; r[3]  = (__bf16)a.w;
  r[4]  = (__bf16)b.x; r[5]  = (__bf16)b.y; r[6]  = (__bf16)b.z; r[7]  = (__bf16)b.w;
  r[8]  = (__bf16)c.x; r[9]  = (__bf16)c.y; r[10] = (__bf16)c.z; r[11] = (__bf16)c.w;
  r[12] = (__bf16)d.x; r[13] = (__bf16)d.y; r[14] = (__bf16)d.z; r[15] = (__bf16)d.w;
  return r;
}

// ---------------------------------------------------------------------------
// Batched GEMM: C[z] = act(A[z] (MxK) @ B[z] (KxN) + bias).
// bsel_lmap!=0 -> B[z] = Bbase + d_LMAP[z]*K*Nc (Wm[LMAP] contraction).
// Block = 128 threads (4 waves), tile 32x64; wave w owns cols [16w,16w+16),
// rows split into two 16-row WMMA tiles sharing one B fragment.
// ---------------------------------------------------------------------------
__global__ void __launch_bounds__(128) gemm_bf16_wmma(
    const float* __restrict__ Abase, long long strideA,
    const float* __restrict__ Bbase, long long strideB, int bsel_lmap,
    const float* __restrict__ bias,
    const float* __restrict__ act_param,
    float* __restrict__ Cbase, long long strideC, int ldc,
    int M, int K, int Nc, int act)
{
  __shared__ __align__(16) float lA[BM][36];    // row-major A tile, pitch 36
  __shared__ __align__(16) float lBt[BN][36];   // TRANSPOSED B tile [col][k]

  const int tid  = threadIdx.x;
  const int wave = tid >> 5;
  const int lane = tid & 31;
  const int z    = blockIdx.z;

  const float* A = Abase + (long long)z * strideA;
  const float* B = bsel_lmap ? (Bbase + (long long)d_LMAP[z] * (long long)K * Nc)
                             : (Bbase + (long long)z * strideB);
  float* C = Cbase + (long long)z * strideC;

  const int row0   = blockIdx.x * BM;
  const int colblk = blockIdx.y * BN;
  const int col0   = colblk + wave * 16;

  v8f accs[2];
#pragma unroll
  for (int t = 0; t < 2; ++t)
#pragma unroll
    for (int i = 0; i < 8; ++i) accs[t][i] = 0.0f;

  for (int k0 = 0; k0 < K; k0 += 32) {
    const bool fullK = (k0 + 32 <= K);
    const bool fastA = fullK && (row0 + BM <= M);
    // ---- A tile (32x32): async DMA / float4 path for interior blocks ------
    if (fastA) {
      for (int i = tid * 4; i < BM * 32; i += 512) {
        int r = i >> 5, kk = i & 31;
#if HAVE_ASYNC_LDS
        async_g2l_16B(&A[(long long)(row0 + r) * K + k0 + kk], &lA[r][kk]);
#else
        float4 v = *(const float4*)&A[(long long)(row0 + r) * K + k0 + kk];
        *(float4*)&lA[r][kk] = v;
#endif
      }
    } else {
      for (int i = tid; i < BM * 32; i += 128) {
        int r = i >> 5, kk = i & 31;
        int gr = row0 + r, gk = k0 + kk;
        lA[r][kk] = (gr < M && gk < K) ? A[(long long)gr * K + gk] : 0.0f;
      }
    }
    // ---- B tile (32x64) stored transposed ---------------------------------
    if (fullK && (colblk + BN <= Nc)) {
      for (int i = tid * 4; i < 32 * BN; i += 512) {
        int kk = i >> 6, c = i & 63;
        float4 v = *(const float4*)&B[(long long)(k0 + kk) * Nc + colblk + c];
        lBt[c + 0][kk] = v.x; lBt[c + 1][kk] = v.y;
        lBt[c + 2][kk] = v.z; lBt[c + 3][kk] = v.w;
      }
    } else {
      for (int i = tid; i < 32 * BN; i += 128) {
        int kk = i >> 6, c = i & 63;
        int gk = k0 + kk, gc = colblk + c;
        lBt[c][kk] = (gk < K && gc < Nc) ? B[(long long)gk * Nc + gc] : 0.0f;
      }
    }
    if (k0 + 32 < K) {  // prefetch next B K-tile -> global_prefetch_b8
      __builtin_prefetch(&B[(long long)(k0 + 32) * Nc + colblk + (tid & 63)], 0, 1);
      __builtin_prefetch(&A[(long long)(row0 + (tid & 31)) * K + k0 + 32], 0, 1);
    }
#if HAVE_ASYNC_LDS
    if (fastA) wait_async0();   // retire this wave's async LDS writes
#endif
    __syncthreads();

    // ---- fragments (contiguous b128 LDS reads) ----------------------------
    // A 16-bit layout: lane m / m+16 holds row m; K runs {khi*8..+7, 16+khi*8..+7}
    const int mrow = lane & 15;
    const int khi  = lane >> 4;
    const float4* pa0 = (const float4*)&lA[mrow][khi * 8];
    const float4* pa1 = (const float4*)&lA[mrow][16 + khi * 8];
    v16bf a0 = pack16(pa0[0], pa0[1], pa1[0], pa1[1]);
    const float4* pb0 = (const float4*)&lA[mrow + 16][khi * 8];
    const float4* pb1 = (const float4*)&lA[mrow + 16][16 + khi * 8];
    v16bf a1 = pack16(pb0[0], pb0[1], pb1[0], pb1[1]);
    // B layout: lane n / n+16 holds col n; K runs {khi*8..+7, 16+khi*8..+7}
    const int bcol = (wave << 4) + (lane & 15);
    const int bk0  = (lane >> 4) * 8;
    const float4* pb = (const float4*)&lBt[bcol][bk0];
    v16bf bf = pack16(pb[0], pb[1], pb[4], pb[5]);

    accs[0] = __builtin_amdgcn_wmma_f32_16x16x32_bf16(
        false, a0, false, bf, (short)0, accs[0], false, false);
    accs[1] = __builtin_amdgcn_wmma_f32_16x16x32_bf16(
        false, a1, false, bf, (short)0, accs[1], false, false);
    __syncthreads();
  }

  // ---- epilogue: bias + activation, guarded stores --------------------------
  const float ap   = act_param ? *act_param : 0.0f;
  const int   gcol = col0 + (lane & 15);
  float bv = 0.0f;
  if (bias && gcol < Nc) bv = bias[gcol];
#pragma unroll
  for (int t = 0; t < 2; ++t) {
#pragma unroll
    for (int r = 0; r < 8; ++r) {
      int grow = row0 + t * 16 + r + ((lane >> 4) << 3);
      if (grow < M && gcol < Nc) {
        float v = accs[t][r] + bv;
        C[(long long)grow * ldc + gcol] = act_apply(v, act, ap);
      }
    }
  }
}

// ---------------------------------------------------------------------------
// Per-edge geometry: direction, spherical harmonics (16), bessel basis (8).
// ---------------------------------------------------------------------------
__global__ void edge_geom_kernel(const float* __restrict__ pos,
                                 const float* __restrict__ shifts,
                                 const int* __restrict__ eidx,
                                 float* __restrict__ sh,
                                 float* __restrict__ ef, int E)
{
  int e = blockIdx.x * blockDim.x + threadIdx.x;
  if (e >= E) return;
  int snd = eidx[e], rcv = eidx[E + e];
  float vx = pos[rcv * 3 + 0] - pos[snd * 3 + 0] + shifts[e * 3 + 0];
  float vy = pos[rcv * 3 + 1] - pos[snd * 3 + 1] + shifts[e * 3 + 1];
  float vz = pos[rcv * 3 + 2] - pos[snd * 3 + 2] + shifts[e * 3 + 2];
  float r = sqrtf(vx * vx + vy * vy + vz * vz) + 1e-9f;
  float x = vx / r, y = vy / r, zc = vz / r;

  const float s3 = 1.7320508075688772f, s15 = 3.872983346207417f;
  const float s5h = 1.118033988749895f, s15h = 1.9364916731037085f;
  const float s358 = 2.091650066335189f, s105 = 10.246950765959598f;
  const float s218 = 1.6201851746019651f, s7h = 1.3228756555322954f;
  const float s105h = 5.123475382979799f;
  float x2 = x * x, y2 = y * y, z2 = zc * zc;
  float* o = sh + (long long)e * 16;
  o[0]  = 1.0f;
  o[1]  = s3 * x;  o[2] = s3 * y;  o[3] = s3 * zc;
  o[4]  = s15 * x * y;  o[5] = s15 * y * zc;
  o[6]  = s5h * (3.0f * z2 - 1.0f);
  o[7]  = s15 * x * zc; o[8] = s15h * (x2 - y2);
  o[9]  = s358 * y * (3.0f * x2 - y2);
  o[10] = s105 * x * y * zc;
  o[11] = s218 * y * (5.0f * z2 - 1.0f);
  o[12] = s7h * (5.0f * z2 * zc - 3.0f * zc);
  o[13] = s218 * x * (5.0f * z2 - 1.0f);
  o[14] = s105h * (x2 - y2) * zc;
  o[15] = s358 * x * (x2 - 3.0f * y2);

  float t = r / 5.0f;
  float t5 = t * t * t * t * t;
  float env = 1.0f - 21.0f * t5 + 35.0f * t5 * t - 15.0f * t5 * t * t;
  env = (t < 1.0f) ? env : 0.0f;
  const float bs = 0.6324555320336759f;   // sqrt(2/RMAX)
  const float pi_r = 3.14159265358979f;
#pragma unroll
  for (int k = 0; k < 8; ++k) {
    ef[(long long)e * 8 + k] = bs * __sinf((k + 1) * pi_r * r / 5.0f) / r * env;
  }
}

// radial layer 1 (K=8, VALU): hid1 = silu(ef @ w1 + b1)
__global__ void radial_l1_kernel(const float* __restrict__ ef,
                                 const float* __restrict__ w1,
                                 const float* __restrict__ b1,
                                 float* __restrict__ hid1, int E)
{
  long long idx = (long long)blockIdx.x * blockDim.x + threadIdx.x;
  if (idx >= (long long)E * 64) return;
  int e = (int)(idx >> 6), f = (int)(idx & 63);
  float acc = b1[f];
#pragma unroll
  for (int k = 0; k < 8; ++k) acc += ef[(long long)e * 8 + k] * w1[k * 64 + f];
  hid1[idx] = acc / (1.0f + __expf(-acc));
}

// one-hot -> species id and embedding row gather: h0 = W_embed[type]
__global__ void node_init_kernel(const float* __restrict__ x,
                                 const float* __restrict__ W_embed,
                                 int* __restrict__ type,
                                 float* __restrict__ h0, int N)
{
  long long idx = (long long)blockIdx.x * blockDim.x + threadIdx.x;
  if (idx >= (long long)N * 64) return;
  int n = (int)(idx >> 6), f = (int)(idx & 63);
  int t = 0; float best = x[(long long)n * 10];
#pragma unroll
  for (int zz = 1; zz < 10; ++zz) {
    float v = x[(long long)n * 10 + zz];
    if (v > best) { best = v; t = zz; }
  }
  if (f == 0) type[n] = t;
  h0[idx] = W_embed[t * 64 + f];
}

// message + segment_sum: At[m][rcv][f] += w[e,f,LMAP[m]] * hs[snd,f] * sh[e,m] / AVG
__global__ void msg_scatter_kernel(const float* __restrict__ wbuf,
                                   const float* __restrict__ hs,
                                   const float* __restrict__ sh,
                                   const int* __restrict__ eidx,
                                   float* __restrict__ At, int E, int N)
{
  int e = blockIdx.x * (blockDim.x >> 5) + (threadIdx.x >> 5);
  if (e >= E) return;
  int lane = threadIdx.x & 31;
  int snd = eidx[e], rcv = eidx[E + e];
  float shv[16];
#pragma unroll
  for (int m = 0; m < 16; ++m) shv[m] = sh[(long long)e * 16 + m] * 0.1f; // /AVG
#pragma unroll
  for (int half = 0; half < 2; ++half) {
    int f = lane + half * 32;
    float h = hs[(long long)snd * 64 + f];
    float w4[4];
#pragma unroll
    for (int l = 0; l < 4; ++l) w4[l] = wbuf[(long long)e * 256 + f * 4 + l];
#pragma unroll
    for (int m = 0; m < 16; ++m) {
      atomicAdd(&At[((long long)m * N + rcv) * 64 + f], w4[d_LMAP[m]] * h * shv[m]);
    }
  }
}

// S2[n,g] = sum_m Bt[m][n][g]^2
__global__ void sq_sum_kernel(const float* __restrict__ Bt, float* __restrict__ S2,
                              long long NF)
{
  long long idx = (long long)blockIdx.x * blockDim.x + threadIdx.x;
  if (idx >= NF) return;
  float acc = 0.0f;
#pragma unroll
  for (int m = 0; m < 16; ++m) { float v = Bt[m * NF + idx]; acc += v * v; }
  S2[idx] = acc;
}

// per-species mix: out[n,f] = sum_g Wa[t,g,f]*Va[n,g] (+Wb,Vb)(+Wc,Vc)
__global__ void species_mix_kernel(const int* __restrict__ type,
                                   const float* __restrict__ Wa, const float* __restrict__ Va,
                                   const float* __restrict__ Wb, const float* __restrict__ Vb,
                                   const float* __restrict__ Wc, const float* __restrict__ Vc,
                                   float* __restrict__ out, int N)
{
  int n = blockIdx.x * (blockDim.x >> 5) + (threadIdx.x >> 5);
  if (n >= N) return;
  int lane = threadIdx.x & 31;
  int t = type[n];
#pragma unroll
  for (int half = 0; half < 2; ++half) {
    int f = lane + half * 32;
    float acc = 0.0f;
    for (int g = 0; g < 64; ++g) {
      long long wi = ((long long)t * 64 + g) * 64 + f;
      acc += Wa[wi] * Va[(long long)n * 64 + g];
      if (Wb) acc += Wb[wi] * Vb[(long long)n * 64 + g];
      if (Wc) acc += Wc[wi] * Vc[(long long)n * 64 + g];
    }
    out[(long long)n * 64 + f] = acc;
  }
}

// split sg (N x 128) -> silu(first half), sigmoid(second half)
__global__ void split_sg_kernel(const float* __restrict__ sg,
                                float* __restrict__ sgA, float* __restrict__ sigB, int N)
{
  long long idx = (long long)blockIdx.x * blockDim.x + threadIdx.x;
  if (idx >= (long long)N * 64) return;
  int n = (int)(idx >> 6), f = (int)(idx & 63);
  float a = sg[(long long)n * 128 + f];
  float b = sg[(long long)n * 128 + 64 + f];
  sgA[idx]  = a / (1.0f + __expf(-a));
  sigB[idx] = 1.0f / (1.0f + __expf(-b));
}

// in-place gate: tv[c][n][f] *= sigB[n][f] for c=0..2
__global__ void gate_vec_kernel(float* __restrict__ tv, const float* __restrict__ sigB,
                                long long NF)
{
  long long idx = (long long)blockIdx.x * blockDim.x + threadIdx.x;
  if (idx >= 3 * NF) return;
  tv[idx] *= sigB[idx % NF];
}

// node = concat(s0[64], v0 reshaped [192: f*3+c], h1[64])
__global__ void build_node_kernel(const float* __restrict__ s0,
                                  const float* __restrict__ v0,
                                  const float* __restrict__ h1,
                                  float* __restrict__ node, int N)
{
  long long idx = (long long)blockIdx.x * blockDim.x + threadIdx.x;
  if (idx >= (long long)N * 64) return;
  int n = (int)(idx >> 6), f = (int)(idx & 63);
  long long NF = (long long)N * 64;
  float* row = node + (long long)n * 320;
  row[f] = s0[idx];
#pragma unroll
  for (int c = 0; c < 3; ++c) row[64 + f * 3 + c] = v0[c * NF + idx];
  row[256 + f] = h1[idx];
}

// out[n,1600+c] = relu'd(stmp) * spd
__global__ void scal_out_kernel(const float* __restrict__ stmp,
                                const float* __restrict__ spd,
                                float* __restrict__ out, int N)
{
  long long idx = (long long)blockIdx.x * blockDim.x + threadIdx.x;
  if (idx >= (long long)N * 4) return;
  int n = (int)(idx >> 2), c = (int)(idx & 3);
  out[(long long)n * OUTC + 1600 + c] = stmp[idx] * spd[idx];
}

// ---------------------------------------------------------------------------
static inline void launch_gemm(const float* A, long long sA,
                               const float* B, long long sB, int lmapsel,
                               const float* bias, const float* actp,
                               float* C, long long sC, int ldc,
                               int M, int K, int Nc, int act, int batch,
                               hipStream_t stream)
{
  dim3 grid((M + BM - 1) / BM, (Nc + BN - 1) / BN, batch);
  gemm_bf16_wmma<<<grid, 128, 0, stream>>>(A, sA, B, sB, lmapsel, bias, actp,
                                           C, sC, ldc, M, K, Nc, act);
}

extern "C" void kernel_launch(void* const* d_in, const int* in_sizes, int n_in,
                              void* d_out, int out_size, void* d_ws, size_t ws_size,
                              hipStream_t stream) {
  const float* x       = (const float*)d_in[0];
  const float* pos     = (const float*)d_in[1];
  const float* shifts  = (const float*)d_in[2];
  const float* spd     = (const float*)d_in[3];
  const int*   eidx    = (const int*)d_in[4];
  const float* W_embed = (const float*)d_in[5];
  const float* W_up0   = (const float*)d_in[6];
  const float* r0_w1   = (const float*)d_in[7];
  const float* r0_b1   = (const float*)d_in[8];
  const float* r0_w2   = (const float*)d_in[9];
  const float* r0_b2   = (const float*)d_in[10];
  const float* r0_w3   = (const float*)d_in[11];
  const float* Wm0     = (const float*)d_in[12];
  const float* Wp0_s   = (const float*)d_in[13];
  const float* Wp0_s2  = (const float*)d_in[14];
  const float* Wp0_v   = (const float*)d_in[15];
  const float* Wr0_sg  = (const float*)d_in[16];
  const float* Wr0_v   = (const float*)d_in[17];
  const float* Wr0_s2  = (const float*)d_in[18];
  const float* Wr0_v2  = (const float*)d_in[19];
  const float* W_up1   = (const float*)d_in[20];
  const float* r1_w1   = (const float*)d_in[21];
  const float* r1_b1   = (const float*)d_in[22];
  const float* r1_w2   = (const float*)d_in[23];
  const float* r1_b2   = (const float*)d_in[24];
  const float* r1_w3   = (const float*)d_in[25];
  const float* Wm1     = (const float*)d_in[26];
  const float* Wsc1    = (const float*)d_in[27];
  const float* Wp1_s   = (const float*)d_in[28];
  const float* Wp1_s2  = (const float*)d_in[29];
  const float* Wr1_1   = (const float*)d_in[30];
  const float* Wr1_2   = (const float*)d_in[31];
  const float* Wd1     = (const float*)d_in[32];
  const float* bd1     = (const float*)d_in[33];
  const float* a_d     = (const float*)d_in[34];
  const float* Wd2     = (const float*)d_in[35];
  const float* bd2     = (const float*)d_in[36];
  const float* Ws1     = (const float*)d_in[37];
  const float* bs1     = (const float*)d_in[38];
  const float* a_s     = (const float*)d_in[39];
  const float* Ws2     = (const float*)d_in[40];
  const float* bs2     = (const float*)d_in[41];

  const int N = in_sizes[1] / 3;   // pos is N x 3
  const int E = in_sizes[4] / 2;   // edge_index is 2 x E
  const long long NF = (long long)N * 64;

  char* ws = (char*)d_ws;
  size_t off = 0;
  auto alloc = [&](size_t bytes) -> void* {
    off = (off + 255) & ~(size_t)255;
    void* p = ws + off; off += bytes; return p;
  };
  int*   type  = (int*)  alloc((size_t)N * 4);
  float* h0    = (float*)alloc(NF * 4);
  float* hsbuf = (float*)alloc(NF * 4);
  float* shb   = (float*)alloc((size_t)E * 16 * 4);
  float* efb   = (float*)alloc((size_t)E * 8 * 4);
  float* hid1  = (float*)alloc((size_t)E * 64 * 4);
  float* hid2  = (float*)alloc((size_t)E * 64 * 4);
  float* wbuf  = (float*)alloc((size_t)E * 256 * 4);
  float* At    = (float*)alloc((size_t)16 * NF * 4);
  float* Bt    = (float*)alloc((size_t)16 * NF * 4);
  float* S2    = (float*)alloc(NF * 4);
  float* ps    = (float*)alloc(NF * 4);           // p0s / p1
  float* pv    = (float*)alloc(3 * NF * 4);       // p0v channels
  float* sgb   = (float*)alloc((size_t)N * 128 * 4);
  float* sgA   = (float*)alloc(NF * 4);
  float* sigB  = (float*)alloc(NF * 4);
  float* s0    = (float*)alloc(NF * 4);
  float* tv    = (float*)alloc(3 * NF * 4);
  float* v0    = (float*)alloc(3 * NF * 4);
  float* h1t   = (float*)alloc(NF * 4);
  float* h1    = (float*)alloc(NF * 4);
  float* node  = (float*)alloc((size_t)N * 320 * 4);
  float* dh    = (float*)alloc((size_t)N * 400 * 4);
  float* sbuf  = (float*)alloc((size_t)N * 100 * 4);
  float* stmp  = (float*)alloc((size_t)N * 4 * 4);
  float* out   = (float*)d_out;

  // --- geometry + embedding ------------------------------------------------
  edge_geom_kernel<<<(E + 255) / 256, 256, 0, stream>>>(pos, shifts, eidx, shb, efb, E);
  node_init_kernel<<<(int)((NF + 255) / 256), 256, 0, stream>>>(x, W_embed, type, h0, N);

  // === conv layer 0 ========================================================
  launch_gemm(h0, 0, W_up0, 0, 0, nullptr, nullptr, hsbuf, 0, 64, N, 64, 64, 0, 1, stream);
  radial_l1_kernel<<<(int)(((long long)E * 64 + 255) / 256), 256, 0, stream>>>(efb, r0_w1, r0_b1, hid1, E);
  launch_gemm(hid1, 0, r0_w2, 0, 0, r0_b2, nullptr, hid2, 0, 64, E, 64, 64, 1, 1, stream);
  launch_gemm(hid2, 0, r0_w3, 0, 0, nullptr, nullptr, wbuf, 0, 256, E, 64, 256, 0, 1, stream);
  (void)hipMemsetAsync(At, 0, (size_t)16 * NF * 4, stream);
  msg_scatter_kernel<<<(E + 7) / 8, 256, 0, stream>>>(wbuf, hsbuf, shb, eidx, At, E, N);
  launch_gemm(At, NF, Wm0, 0, 1, nullptr, nullptr, Bt, NF, 64, N, 64, 64, 0, 16, stream);

  sq_sum_kernel<<<(int)((NF + 255) / 256), 256, 0, stream>>>(Bt, S2, NF);
  species_mix_kernel<<<(N + 7) / 8, 256, 0, stream>>>(type, Wp0_s, Bt, Wp0_s2, S2,
                                                      nullptr, nullptr, ps, N);
  for (int c = 0; c < 3; ++c)
    species_mix_kernel<<<(N + 7) / 8, 256, 0, stream>>>(type, Wp0_v, Bt + (1 + c) * NF,
                                                        nullptr, nullptr, nullptr, nullptr,
                                                        pv + c * NF, N);
  launch_gemm(ps, 0, Wr0_sg, 0, 0, nullptr, nullptr, sgb, 0, 128, N, 64, 128, 0, 1, stream);
  split_sg_kernel<<<(int)((NF + 255) / 256), 256, 0, stream>>>(sgb, sgA, sigB, N);
  launch_gemm(sgA, 0, Wr0_s2, 0, 0, nullptr, nullptr, s0, 0, 64, N, 64, 64, 0, 1, stream);
  launch_gemm(pv, NF, Wr0_v, 0, 0, nullptr, nullptr, tv, NF, 64, N, 64, 64, 0, 3, stream);
  gate_vec_kernel<<<(int)((3 * NF + 255) / 256), 256, 0, stream>>>(tv, sigB, NF);
  launch_gemm(tv, NF, Wr0_v2, 0, 0, nullptr, nullptr, v0, NF, 64, N, 64, 64, 0, 3, stream);

  // === conv layer 1 ========================================================
  launch_gemm(s0, 0, W_up1, 0, 0, nullptr, nullptr, hsbuf, 0, 64, N, 64, 64, 0, 1, stream);
  radial_l1_kernel<<<(int)(((long long)E * 64 + 255) / 256), 256, 0, stream>>>(efb, r1_w1, r1_b1, hid1, E);
  launch_gemm(hid1, 0, r1_w2, 0, 0, r1_b2, nullptr, hid2, 0, 64, E, 64, 64, 1, 1, stream);
  launch_gemm(hid2, 0, r1_w3, 0, 0, nullptr, nullptr, wbuf, 0, 256, E, 64, 256, 0, 1, stream);
  (void)hipMemsetAsync(At, 0, (size_t)16 * NF * 4, stream);
  msg_scatter_kernel<<<(E + 7) / 8, 256, 0, stream>>>(wbuf, hsbuf, shb, eidx, At, E, N);
  launch_gemm(At, NF, Wm1, 0, 1, nullptr, nullptr, Bt, NF, 64, N, 64, 64, 0, 16, stream);

  sq_sum_kernel<<<(int)((NF + 255) / 256), 256, 0, stream>>>(Bt, S2, NF);
  species_mix_kernel<<<(N + 7) / 8, 256, 0, stream>>>(type, Wp1_s, Bt, Wp1_s2, S2,
                                                      Wsc1, s0, ps, N);
  launch_gemm(ps, 0, Wr1_1, 0, 0, nullptr, nullptr, h1t, 0, 64, N, 64, 64, 1, 1, stream);
  launch_gemm(h1t, 0, Wr1_2, 0, 0, nullptr, nullptr, h1, 0, 64, N, 64, 64, 0, 1, stream);

  // === readout =============================================================
  build_node_kernel<<<(int)((NF + 255) / 256), 256, 0, stream>>>(s0, v0, h1, node, N);
  launch_gemm(node, 0, Wd1, 0, 0, bd1, a_d, dh, 0, 400, N, 320, 400, 4, 1, stream);
  launch_gemm(dh, 0, Wd2, 0, 0, bd2, nullptr, out, 0, OUTC, N, 400, 1600, 3, 1, stream);
  launch_gemm(node, 0, Ws1, 0, 0, bs1, a_s, sbuf, 0, 100, N, 320, 100, 4, 1, stream);
  launch_gemm(sbuf, 0, Ws2, 0, 0, bs2, nullptr, stmp, 0, 4, N, 100, 4, 5, 1, stream);
  scal_out_kernel<<<(int)(((long long)N * 4 + 255) / 256), 256, 0, stream>>>(stmp, spd, out, N);
  (void)n_in; (void)in_sizes; (void)out_size; (void)ws_size;
}
